// CrossAttention_35175782154388
// MI455X (gfx1250) — matrix operand order repaired and checked by reading, not compiled, not run on hardware
//
// MI455X (gfx1250, CDNA5) cross-attention: all contractions via
// v_wmma_f32_16x16x32_bf16 (f32 accumulate), softmax in f32.
//
// Pipeline:
//   1) cvt f32->bf16: query, input_flatten (=V), Wq, Wk, Wo
//   2) Q = q_b @ Wq^T + bq          (gemm_bt, bf16 out)
//   3) K = x_b @ Wk^T + bk          (gemm_bt, bf16 out)
//   4) per batch n:
//        S = Q_n @ K_n^T            (gemm_bt, f32 out)
//        P = softmax(mask(S))       (row kernel, bf16 out)
//        O_n = P @ V_n              (gemm_nn, bf16 out)
//   5) out = O @ Wo^T + bo          (gemm_bt, f32 out -> d_out)
//
// GEMMs use double-buffered LDS tiles; when the toolchain exposes the CDNA5
// async-copy builtins, tile staging is GLOBAL_LOAD_ASYNC_TO_LDS_B128
// (ASYNCcnt) fully overlapped with the WMMAs of the current tile.
// Requires ws_size >= ~220 MB.

#include <hip/hip_runtime.h>
#include <stdint.h>

typedef __attribute__((ext_vector_type(16))) __bf16   v16bf;
typedef __attribute__((ext_vector_type(8)))  float    v8f;
typedef __attribute__((ext_vector_type(8)))  uint32_t v8u;
typedef int v4i_vs __attribute__((vector_size(16)));   // builtin's int4 type

#if defined(__AMDGCN__) && __has_builtin(__builtin_amdgcn_global_load_async_to_lds_b128) && \
    __has_builtin(__builtin_amdgcn_s_wait_asynccnt)
#define USE_ASYNC_LDS 1
#else
#define USE_ASYNC_LDS 0
#endif

__device__ __forceinline__ uint16_t f32_to_bf16(float f) {
  uint32_t u = __builtin_bit_cast(uint32_t, f);
  u += 0x7FFFu + ((u >> 16) & 1u);          // round-to-nearest-even
  return (uint16_t)(u >> 16);
}

// Copy 32 B (8 dwords) global -> LDS for this lane.
__device__ __forceinline__ void stage32B(uint32_t* l, const uint32_t* g) {
#if USE_ASYNC_LDS
  __builtin_amdgcn_global_load_async_to_lds_b128(
      (__attribute__((address_space(1))) v4i_vs*)g,
      (__attribute__((address_space(3))) v4i_vs*)l, 0, 0);
  __builtin_amdgcn_global_load_async_to_lds_b128(
      (__attribute__((address_space(1))) v4i_vs*)(g + 4),
      (__attribute__((address_space(3))) v4i_vs*)(l + 4), 0, 0);
#else
#pragma unroll
  for (int j = 0; j < 8; ++j) l[j] = g[j];
#endif
}

__device__ __forceinline__ void stage_wait() {
#if USE_ASYNC_LDS
  __builtin_amdgcn_s_wait_asynccnt(0);
#endif
}

// ---------------------------------------------------------------- conversion
__global__ __launch_bounds__(256) void cvt_f32_to_bf16(
    const float* __restrict__ in, uint16_t* __restrict__ out, int n) {
  int i = (blockIdx.x * 256 + threadIdx.x) * 4;
  if (i >= n) return;
  float4 f = *(const float4*)(in + i);
  uint2 o;
  o.x = (uint32_t)f32_to_bf16(f.x) | ((uint32_t)f32_to_bf16(f.y) << 16);
  o.y = (uint32_t)f32_to_bf16(f.z) | ((uint32_t)f32_to_bf16(f.w) << 16);
  *(uint2*)(out + i) = o;
}

// ------------------------------------------------------- WMMA fragment loads
// LDS tiles are 128 rows x 32 bf16, padded to 20 dwords/row (80 B):
// 16-B aligned fragment chunks, and 16 consecutive rows x 4 dwords cover all
// 64 LDS banks exactly once (conflict-free ds_load_b128).
#define LDSW 20
#define TBUF (128 * LDSW)   // dwords per tile buffer

// A matrix 16x32 bf16 (ISA 7.12.2): lanes 0-15 (g=0) K=0..7 & 16..23,
// lanes 16-31 (g=1) K=8..15 & 24..31, two bf16 per VGPR.
__device__ __forceinline__ v16bf frag_a(const uint32_t* lds, int row, int g) {
  const uint32_t* p = lds + row * LDSW + g * 4;
  v8u u;
#pragma unroll
  for (int j = 0; j < 4; ++j) { u[j] = p[j]; u[4 + j] = p[8 + j]; }
  return __builtin_bit_cast(v16bf, u);
}

// B matrix 32x16 bf16: lane holds column n = lane&15; lanes 0-15 carry
// K=0..7 & 16..23, lanes 16-31 carry K=8..15 & 24..31 (same as A).
__device__ __forceinline__ v16bf frag_b(const uint32_t* lds, int row, int g) {
  const uint32_t* p = lds + row * LDSW + g * 4;
  v8u u;
#pragma unroll
  for (int j = 0; j < 4; ++j) { u[j] = p[j]; u[4 + j] = p[8 + j]; }
  return __builtin_bit_cast(v16bf, u);
}

// --------------------------------------------------- C = A(MxK) @ B(NxK)^T
// Both operands row-major along K (torch-Linear x@W^T and Q@K^T pattern).
// Block: 256 thr = 8 waves (2x4), block tile 128x128, wave tile 64x32.
// Double-buffered: stage tile i+1 (async) while running WMMAs on tile i.
template <bool OUT_BF16, bool HAS_BIAS>
__global__ __launch_bounds__(256) void gemm_bt(
    const uint16_t* __restrict__ A, const uint16_t* __restrict__ B,
    const float* __restrict__ bias, void* __restrict__ C,
    int M, int N, int K) {
  __shared__ uint32_t As[2 * TBUF];
  __shared__ uint32_t Bs[2 * TBUF];
  const int tid = threadIdx.x;
  const int lane = tid & 31, wave = tid >> 5;
  const int m0 = (wave >> 2) * 64, n0 = (wave & 3) * 32;
  const int lr = lane & 15, g = lane >> 4;
  const size_t Kd = (size_t)(K >> 1);

  v8f acc[4][2];
#pragma unroll
  for (int i = 0; i < 4; ++i)
#pragma unroll
    for (int j = 0; j < 2; ++j)
#pragma unroll
      for (int e = 0; e < 8; ++e) acc[i][j][e] = 0.f;

  const int srow = tid >> 1, shalf = (tid & 1) * 8;
  const int soff = srow * LDSW + shalf;
  const uint32_t* gA = (const uint32_t*)A + ((size_t)blockIdx.y * 128 + srow) * Kd + shalf;
  const uint32_t* gB = (const uint32_t*)B + ((size_t)blockIdx.x * 128 + srow) * Kd + shalf;

  const int nIter = K >> 5;
  // prologue: tile 0 -> buffer 0
  stage32B(As + soff, gA);
  stage32B(Bs + soff, gB);
  stage_wait();
  __syncthreads();

  for (int it = 0; it < nIter; ++it) {
    const int c = it & 1;
    if (it + 1 < nIter) {                       // overlap copy of tile it+1
      stage32B(As + (c ^ 1) * TBUF + soff, gA + (size_t)16 * (it + 1));
      stage32B(Bs + (c ^ 1) * TBUF + soff, gB + (size_t)16 * (it + 1));
    }
    if (it + 2 < nIter) {                       // pull tile it+2 toward GL2
      __builtin_prefetch(gA + (size_t)16 * (it + 2), 0, 0);
      __builtin_prefetch(gB + (size_t)16 * (it + 2), 0, 0);
    }

    const uint32_t* aT = As + c * TBUF;
    const uint32_t* bT = Bs + c * TBUF;
    v16bf af[4], bfr[2];
#pragma unroll
    for (int i = 0; i < 4; ++i) af[i] = frag_a(aT, m0 + i * 16 + lr, g);
#pragma unroll
    for (int j = 0; j < 2; ++j) bfr[j] = frag_b(bT, n0 + j * 16 + lr, g);
#pragma unroll
    for (int i = 0; i < 4; ++i)
#pragma unroll
      for (int j = 0; j < 2; ++j)
        acc[i][j] = __builtin_amdgcn_wmma_f32_16x16x32_bf16(
            false, af[i], false, bfr[j], (short)0, acc[i][j], false, false);

    if (it + 1 < nIter) {
      stage_wait();
      __syncthreads();
    }
  }

#pragma unroll
  for (int i = 0; i < 4; ++i)
#pragma unroll
    for (int j = 0; j < 2; ++j) {
      const int col = blockIdx.x * 128 + n0 + j * 16 + lr;
      const float bv = HAS_BIAS ? bias[col] : 0.f;
#pragma unroll
      for (int r = 0; r < 8; ++r) {
        const int row = blockIdx.y * 128 + m0 + i * 16 + g * 8 + r;
        const float v = acc[i][j][r] + bv;
        if (OUT_BF16)
          ((uint16_t*)C)[(size_t)row * N + col] = f32_to_bf16(v);
        else
          ((float*)C)[(size_t)row * N + col] = v;
      }
    }
  (void)M;
}

// ----------------------------------------------------- C = A(MxK) @ B(KxN)
// P@V pattern: V tile (32k x 128n) transposed into LDS as [n][k] so the
// B-fragment read is identical to gemm_bt. A tile staged async; V tile goes
// through VGPRs (transpose), scattered into the *other* buffer after the
// WMMAs so its global loads also overlap compute.
template <bool OUT_BF16, bool HAS_BIAS>
__global__ __launch_bounds__(256) void gemm_nn(
    const uint16_t* __restrict__ A, const uint16_t* __restrict__ B,
    const float* __restrict__ bias, void* __restrict__ C,
    int M, int N, int K) {
  __shared__ uint32_t As[2 * TBUF];
  __shared__ uint32_t Bs[2 * TBUF];
  const int tid = threadIdx.x;
  const int lane = tid & 31, wave = tid >> 5;
  const int m0 = (wave >> 2) * 64, n0 = (wave & 3) * 32;
  const int lr = lane & 15, g = lane >> 4;
  const size_t Kd = (size_t)(K >> 1);
  const size_t Nd = (size_t)(N >> 1);

  v8f acc[4][2];
#pragma unroll
  for (int i = 0; i < 4; ++i)
#pragma unroll
    for (int j = 0; j < 2; ++j)
#pragma unroll
      for (int e = 0; e < 8; ++e) acc[i][j][e] = 0.f;

  const int srow = tid >> 1, shalf = (tid & 1) * 8;
  const int soff = srow * LDSW + shalf;
  const uint32_t* gA = (const uint32_t*)A + ((size_t)blockIdx.y * 128 + srow) * Kd + shalf;

  const int bkr = tid >> 3;            // k-row within tile, 0..31
  const int bn0 = (tid & 7) * 16;      // n offset, steps of 16
  const uint32_t* gB = (const uint32_t*)B + (size_t)bkr * Nd +
                       (((size_t)blockIdx.x * 128 + bn0) >> 1);

  auto scatterB = [&](uint32_t* buf, const uint32_t* t0) {
    uint16_t* b16 = (uint16_t*)buf;
#pragma unroll
    for (int e = 0; e < 16; ++e) {     // 16-bit scatter transpose into LDS
      const uint32_t d = t0[e >> 1];
      b16[(bn0 + e) * (2 * LDSW) + bkr] =
          (uint16_t)((e & 1) ? (d >> 16) : (d & 0xffffu));
    }
  };

  const int nIter = K >> 5;
  // prologue: tile 0 -> buffer 0
  stage32B(As + soff, gA);
  {
    uint32_t t0[8];
#pragma unroll
    for (int j = 0; j < 8; ++j) t0[j] = gB[j];
    scatterB(Bs, t0);
  }
  stage_wait();
  __syncthreads();

  for (int it = 0; it < nIter; ++it) {
    const int c = it & 1;
    uint32_t t0[8];
    if (it + 1 < nIter) {
      stage32B(As + (c ^ 1) * TBUF + soff, gA + (size_t)16 * (it + 1));
      const uint32_t* gBn = gB + (size_t)32 * Nd * (it + 1);
#pragma unroll
      for (int j = 0; j < 8; ++j) t0[j] = gBn[j];   // V tile -> VGPRs
    }

    const uint32_t* aT = As + c * TBUF;
    const uint32_t* bT = Bs + c * TBUF;
    v16bf af[4], bfr[2];
#pragma unroll
    for (int i = 0; i < 4; ++i) af[i] = frag_a(aT, m0 + i * 16 + lr, g);
#pragma unroll
    for (int j = 0; j < 2; ++j) bfr[j] = frag_b(bT, n0 + j * 16 + lr, g);
#pragma unroll
    for (int i = 0; i < 4; ++i)
#pragma unroll
      for (int j = 0; j < 2; ++j)
        acc[i][j] = __builtin_amdgcn_wmma_f32_16x16x32_bf16(
            false, af[i], false, bfr[j], (short)0, acc[i][j], false, false);

    if (it + 1 < nIter) {
      scatterB(Bs + (c ^ 1) * TBUF, t0);
      stage_wait();
      __syncthreads();
    }
  }

#pragma unroll
  for (int i = 0; i < 4; ++i)
#pragma unroll
    for (int j = 0; j < 2; ++j) {
      const int col = blockIdx.x * 128 + n0 + j * 16 + lr;
      const float bv = HAS_BIAS ? bias[col] : 0.f;
#pragma unroll
      for (int r = 0; r < 8; ++r) {
        const int row = blockIdx.y * 128 + m0 + i * 16 + g * 8 + r;
        const float v = acc[i][j][r] + bv;
        if (OUT_BF16)
          ((uint16_t*)C)[(size_t)row * N + col] = f32_to_bf16(v);
        else
          ((float*)C)[(size_t)row * N + col] = v;
      }
    }
  (void)M;
}

// ----------------------------------------- masked row softmax (4096 cols)
// One block per score row; padding mask (nonzero => -inf) applied on load.
__global__ __launch_bounds__(256) void softmax_rows(
    const float* __restrict__ S, const int* __restrict__ mask,
    uint16_t* __restrict__ P) {
  const int row = blockIdx.x, tid = threadIdx.x;
  const float* s = S + (size_t)row * 4096;
  uint16_t* p = P + (size_t)row * 4096;
  float v[16];
  float mx = -INFINITY;
#pragma unroll
  for (int i = 0; i < 16; ++i) {
    const int l = i * 256 + tid;
    const float val = mask[l] ? -INFINITY : s[l];
    v[i] = val;
    mx = fmaxf(mx, val);
  }
#pragma unroll
  for (int off = 16; off > 0; off >>= 1) mx = fmaxf(mx, __shfl_xor(mx, off, 32));
  __shared__ float red[8];
  const int wave = tid >> 5;
  if ((tid & 31) == 0) red[wave] = mx;
  __syncthreads();
  mx = red[0];
#pragma unroll
  for (int w = 1; w < 8; ++w) mx = fmaxf(mx, red[w]);
  __syncthreads();
  float sum = 0.f;
#pragma unroll
  for (int i = 0; i < 16; ++i) { const float e = __expf(v[i] - mx); v[i] = e; sum += e; }
#pragma unroll
  for (int off = 16; off > 0; off >>= 1) sum += __shfl_xor(sum, off, 32);
  if ((tid & 31) == 0) red[wave] = sum;
  __syncthreads();
  sum = 0.f;
#pragma unroll
  for (int w = 0; w < 8; ++w) sum += red[w];
  const float inv = 1.0f / sum;
#pragma unroll
  for (int i = 0; i < 16; ++i) p[i * 256 + tid] = f32_to_bf16(v[i] * inv);
}

// -------------------------------------------------------------------- launch
extern "C" void kernel_launch(void* const* d_in, const int* in_sizes, int n_in,
                              void* d_out, int out_size, void* d_ws, size_t ws_size,
                              hipStream_t stream) {
  constexpr int NB = 8, LQ = 1024, LIN = 4096, DM = 1024;
  const float* query = (const float*)d_in[0];
  const float* xin   = (const float*)d_in[1];
  const int*   mask  = (const int*)d_in[2];   // bool mask uploaded as int32
  const float* Wq    = (const float*)d_in[3];
  const float* bq    = (const float*)d_in[4];
  const float* Wk    = (const float*)d_in[5];
  const float* bk    = (const float*)d_in[6];
  const float* Wo    = (const float*)d_in[7];
  const float* bo    = (const float*)d_in[8];

  char* ws = (char*)d_ws;
  auto take = [&](size_t bytes) -> char* {
    char* p = ws;
    ws += (bytes + 255) & ~(size_t)255;
    return p;
  };
  uint16_t* q_b  = (uint16_t*)take((size_t)NB * LQ * DM * 2);
  uint16_t* x_b  = (uint16_t*)take((size_t)NB * LIN * DM * 2);   // K input & V
  uint16_t* wq_b = (uint16_t*)take((size_t)DM * DM * 2);
  uint16_t* wk_b = (uint16_t*)take((size_t)DM * DM * 2);
  uint16_t* wo_b = (uint16_t*)take((size_t)DM * DM * 2);
  uint16_t* Qp   = (uint16_t*)take((size_t)NB * LQ * DM * 2);
  uint16_t* Kp   = (uint16_t*)take((size_t)NB * LIN * DM * 2);
  float*    Sb   = (float*)take((size_t)LQ * LIN * 4);           // per-batch
  uint16_t* Pb   = (uint16_t*)take((size_t)LQ * LIN * 2);        // per-batch
  uint16_t* Opre = (uint16_t*)take((size_t)NB * LQ * DM * 2);

  auto cvt = [&](const float* in, uint16_t* out, size_t n) {
    cvt_f32_to_bf16<<<(int)((n / 4 + 255) / 256), 256, 0, stream>>>(in, out, (int)n);
  };
  cvt(query, q_b, (size_t)NB * LQ * DM);
  cvt(xin,   x_b, (size_t)NB * LIN * DM);
  cvt(Wq, wq_b, (size_t)DM * DM);
  cvt(Wk, wk_b, (size_t)DM * DM);
  cvt(Wo, wo_b, (size_t)DM * DM);

  gemm_bt<true, true><<<dim3(DM / 128, NB * LQ / 128), 256, 0, stream>>>(
      q_b, wq_b, bq, Qp, NB * LQ, DM, DM);
  gemm_bt<true, true><<<dim3(DM / 128, NB * LIN / 128), 256, 0, stream>>>(
      x_b, wk_b, bk, Kp, NB * LIN, DM, DM);

  for (int n = 0; n < NB; ++n) {
    gemm_bt<false, false><<<dim3(LIN / 128, LQ / 128), 256, 0, stream>>>(
        Qp + (size_t)n * LQ * DM, Kp + (size_t)n * LIN * DM, nullptr,
        Sb, LQ, LIN, DM);
    softmax_rows<<<LQ, 256, 0, stream>>>(Sb, mask + (size_t)n * LIN, Pb);
    gemm_nn<true, false><<<dim3(DM / 128, LQ / 128), 256, 0, stream>>>(
        Pb, x_b + (size_t)n * LIN * DM, nullptr,
        Opre + (size_t)n * LQ * DM, LQ, DM, LIN);
  }

  gemm_bt<false, true><<<dim3(DM / 128, NB * LQ / 128), 256, 0, stream>>>(
      Opre, wo_b, bo, d_out, NB * LQ, DM, DM);

  (void)in_sizes; (void)n_in; (void)out_size; (void)ws_size;
}
